// LSTM_cell_81999515615465
// MI455X (gfx1250) — matrix operand order repaired and checked by reading, not compiled, not run on hardware
//
#include <hip/hip_runtime.h>
#include <hip/hip_bf16.h>

// ---------------------------------------------------------------------------
// Sememe-LSTM for gfx1250 (MI455X). All GEMMs run through
// v_wmma_f32_16x16x32_bf16 (bf16 inputs, f32 accumulation).
// ---------------------------------------------------------------------------

#define T_DIM 256
#define B_DIM 64
#define I_DIM 512
#define M_DIM 512
#define K_DIM 512                 // every GEMM in this model has K = 512
#define PCOLS 2560                // P columns: [0,1536) iou | [1536,2048) f | [2048,2560) fs
#define ROWS_X (T_DIM * B_DIM)    // 16384

typedef __attribute__((ext_vector_type(16))) __bf16        v16bf;
typedef __attribute__((ext_vector_type(8)))  float         v8f;
typedef __attribute__((ext_vector_type(4)))  unsigned int  uint4v;

__device__ __forceinline__ unsigned short f2bf(float f) {
    // round-to-nearest-even f32 -> bf16
    unsigned int u = __float_as_uint(f);
    unsigned int r = u + 0x7FFFu + ((u >> 16) & 1u);
    return (unsigned short)(r >> 16);
}

__device__ __forceinline__ float sigm(float x) {
    return 1.0f / (1.0f + __expf(-x));
}

// ---------------------------------------------------------------------------
// f32 -> bf16 conversion
// ---------------------------------------------------------------------------
__global__ void k_f2bf(const float* __restrict__ src,
                       unsigned short* __restrict__ dst, int n) {
    int i = blockIdx.x * blockDim.x + threadIdx.x;
    if (i < n) dst[i] = f2bf(src[i]);
}

// ---------------------------------------------------------------------------
// Initialize P with the (pre-summed) biases.
// cols [0,1536):  b_ioux + b_iouh + b_ious
// cols [1536,2048): b_fx + b_fh
// cols [2048,2560): b_fxs + b_fs
// ---------------------------------------------------------------------------
__global__ void k_init_P(float* __restrict__ P,
                         const float* __restrict__ b_ioux,
                         const float* __restrict__ b_iouh,
                         const float* __restrict__ b_ious,
                         const float* __restrict__ b_fx,
                         const float* __restrict__ b_fh,
                         const float* __restrict__ b_fxs,
                         const float* __restrict__ b_fs) {
    size_t idx = (size_t)blockIdx.x * blockDim.x + threadIdx.x;
    const size_t total = (size_t)ROWS_X * PCOLS;
    if (idx >= total) return;
    int c = (int)(idx % PCOLS);
    float v;
    if (c < 1536)      v = b_ioux[c] + b_iouh[c] + b_ious[c];
    else if (c < 2048) v = b_fx[c - 1536] + b_fh[c - 1536];
    else               v = b_fxs[c - 2048] + b_fs[c - 2048];
    P[idx] = v;
}

// ---------------------------------------------------------------------------
// Generic WMMA bf16 GEMM:
//   C[row, colOff + n] (accum ? += : =) sum_k A[row,k] * W[n,k]
// A: bf16 [rows x 512] row-major (K contiguous)
// W: bf16 [N x 512]    row-major (K contiguous)  -> row n of W is output col n
// Grid: blockIdx.x over rows/64, blockIdx.y over N/64. Block = 128 thr = 4 waves.
// Each wave computes a 16(row) x 64(col) strip: 4 accumulators, K-loop of 16x32.
// ---------------------------------------------------------------------------
__global__ __launch_bounds__(128) void k_gemm_bf16(
    const unsigned short* __restrict__ A,
    const unsigned short* __restrict__ W,
    float* __restrict__ C,
    int ldc, int colOff, int accum) {

    const int lane = threadIdx.x & 31;
    const int wv   = threadIdx.x >> 5;
    const int hh   = lane >> 4;     // half-wave: 0 -> lanes 0-15, 1 -> lanes 16-31
    const int r    = lane & 15;
    const int row0 = blockIdx.x * 64 + wv * 16;
    const int col0 = blockIdx.y * 64;

    v8f acc0 = {}, acc1 = {}, acc2 = {}, acc3 = {};

    // A fragment (16x32 bf16, MxK): lane r / r+16 holds row M=r.
    //   elems 0..15 -> K = k0 + hh*16 + e
    const unsigned short* aRow = A + (size_t)(row0 + r) * K_DIM + hh * 16;
    // B fragment (32x16 bf16, KxN): lane holds column n=r; elems contiguous in K.
    //   elems 0..15 -> K = k0 + hh*16 + e
    const unsigned short* w0 = W + (size_t)(col0 +  0 + r) * K_DIM + hh * 16;
    const unsigned short* w1 = W + (size_t)(col0 + 16 + r) * K_DIM + hh * 16;
    const unsigned short* w2 = W + (size_t)(col0 + 32 + r) * K_DIM + hh * 16;
    const unsigned short* w3 = W + (size_t)(col0 + 48 + r) * K_DIM + hh * 16;

    for (int k0 = 0; k0 < K_DIM; k0 += 32) {
        union { uint4v u[2]; v16bf v; } af, b0, b1, b2, b3;
        af.u[0] = *(const uint4v*)(aRow + k0);
        af.u[1] = *(const uint4v*)(aRow + k0 + 8);
        b0.u[0] = *(const uint4v*)(w0 + k0);  b0.u[1] = *(const uint4v*)(w0 + k0 + 8);
        b1.u[0] = *(const uint4v*)(w1 + k0);  b1.u[1] = *(const uint4v*)(w1 + k0 + 8);
        b2.u[0] = *(const uint4v*)(w2 + k0);  b2.u[1] = *(const uint4v*)(w2 + k0 + 8);
        b3.u[0] = *(const uint4v*)(w3 + k0);  b3.u[1] = *(const uint4v*)(w3 + k0 + 8);

        acc0 = __builtin_amdgcn_wmma_f32_16x16x32_bf16(false, af.v, false, b0.v,
                                                       (short)0, acc0, false, false);
        acc1 = __builtin_amdgcn_wmma_f32_16x16x32_bf16(false, af.v, false, b1.v,
                                                       (short)0, acc1, false, false);
        acc2 = __builtin_amdgcn_wmma_f32_16x16x32_bf16(false, af.v, false, b2.v,
                                                       (short)0, acc2, false, false);
        acc3 = __builtin_amdgcn_wmma_f32_16x16x32_bf16(false, af.v, false, b3.v,
                                                       (short)0, acc3, false, false);
    }

    // C/D layout: VGPR v, lanes 0-15 -> row v, lanes 16-31 -> row v+8; col = lane&15.
    float* cr = C + (size_t)(row0 + hh * 8) * ldc + colOff + col0 + r;
    if (accum) {
#pragma unroll
        for (int v = 0; v < 8; ++v) {
            cr[(size_t)v * ldc +  0] += acc0[v];
            cr[(size_t)v * ldc + 16] += acc1[v];
            cr[(size_t)v * ldc + 32] += acc2[v];
            cr[(size_t)v * ldc + 48] += acc3[v];
        }
    } else {
#pragma unroll
        for (int v = 0; v < 8; ++v) {
            cr[(size_t)v * ldc +  0] = acc0[v];
            cr[(size_t)v * ldc + 16] = acc1[v];
            cr[(size_t)v * ldc + 32] = acc2[v];
            cr[(size_t)v * ldc + 48] = acc3[v];
        }
    }
}

// ---------------------------------------------------------------------------
// Per-step fused gates:  [B=64, M=512] elementwise
//   i = sig(P[.,m]     + tmp[.,m])
//   o = sig(P[.,512+m] + tmp[.,512+m])
//   u = tanh(P[.,1024+m] + tmp[.,1024+m])
//   f = sig(P[.,1536+m] + tmp[.,1536+m])
//   fs = sig(P[.,2048+m])                       (fully precomputed linear part)
//   c' = i*u + f*c + fs*sememe_c[t]
//   h' = o*tanh(c')
// Writes h' (f32) to output, h' (bf16) for the next step's GEMM, updates c.
// ---------------------------------------------------------------------------
__global__ void k_gates(const float* __restrict__ P,
                        const float* __restrict__ tmp,
                        float* __restrict__ c,
                        const float* __restrict__ sememe_c,
                        float* __restrict__ out,
                        unsigned short* __restrict__ hbf,
                        int t) {
    int idx = blockIdx.x * blockDim.x + threadIdx.x;   // < 64*512
    int b = idx >> 9;
    int m = idx & 511;
    const float* Prow = P + (size_t)(t * B_DIM + b) * PCOLS;
    const float* trow = tmp + (size_t)b * 2048;

    float gi = sigm(Prow[m]        + trow[m]);
    float go = sigm(Prow[512 + m]  + trow[512 + m]);
    float gu = tanhf(Prow[1024 + m] + trow[1024 + m]);
    float gf = sigm(Prow[1536 + m] + trow[1536 + m]);
    float gfs = sigm(Prow[2048 + m]);

    float sc = sememe_c[(size_t)(t * B_DIM + b) * M_DIM + m];
    float cn = gi * gu + gf * c[idx] + gfs * sc;
    float hn = go * tanhf(cn);

    c[idx] = cn;
    out[(size_t)t * (B_DIM * M_DIM) + idx] = hn;
    hbf[idx] = f2bf(hn);
    if (t == T_DIM - 1) {
        const size_t base = (size_t)T_DIM * B_DIM * M_DIM;
        out[base + idx] = cn;                      // c_T
        out[base + (size_t)B_DIM * M_DIM + idx] = hn;  // h_T
    }
}

// ---------------------------------------------------------------------------
extern "C" void kernel_launch(void* const* d_in, const int* in_sizes, int n_in,
                              void* d_out, int out_size, void* d_ws, size_t ws_size,
                              hipStream_t stream) {
    (void)in_sizes; (void)n_in; (void)out_size; (void)ws_size;

    const float* inputs   = (const float*)d_in[0];
    const float* sememe_c = (const float*)d_in[1];
    const float* sememe_h = (const float*)d_in[2];
    const float* c0       = (const float*)d_in[3];
    const float* h0       = (const float*)d_in[4];
    const float* W_ioux   = (const float*)d_in[5];
    const float* b_ioux   = (const float*)d_in[6];
    const float* W_iouh   = (const float*)d_in[7];
    const float* b_iouh   = (const float*)d_in[8];
    const float* W_ious   = (const float*)d_in[9];
    const float* b_ious   = (const float*)d_in[10];
    const float* W_fx     = (const float*)d_in[11];
    const float* b_fx     = (const float*)d_in[12];
    const float* W_fxs    = (const float*)d_in[13];
    const float* b_fxs    = (const float*)d_in[14];
    const float* W_fh     = (const float*)d_in[15];
    const float* b_fh     = (const float*)d_in[16];
    const float* W_fs     = (const float*)d_in[17];
    const float* b_fs     = (const float*)d_in[18];
    float* out = (float*)d_out;

    // ---- workspace layout (all region sizes are multiples of 256 B) ----
    char* ws = (char*)d_ws;
    size_t off = 0;
    float* P = (float*)(ws + off);            off += (size_t)ROWS_X * PCOLS * 4;   // 167.8 MB
    unsigned short* Xbf    = (unsigned short*)(ws + off); off += (size_t)ROWS_X * I_DIM * 2;
    unsigned short* Sbf    = (unsigned short*)(ws + off); off += (size_t)ROWS_X * M_DIM * 2;
    unsigned short* Wioux  = (unsigned short*)(ws + off); off += 1536 * 512 * 2;
    unsigned short* Wfx    = (unsigned short*)(ws + off); off += 512 * 512 * 2;
    unsigned short* Wfxs   = (unsigned short*)(ws + off); off += 512 * 512 * 2;
    unsigned short* Wious  = (unsigned short*)(ws + off); off += 1536 * 512 * 2;
    unsigned short* Wfs    = (unsigned short*)(ws + off); off += 512 * 512 * 2;
    unsigned short* Whcat  = (unsigned short*)(ws + off); off += 2048 * 512 * 2;   // [W_iouh; W_fh]
    unsigned short* Hbf    = (unsigned short*)(ws + off); off += B_DIM * M_DIM * 2;
    float* Cst             = (float*)(ws + off);          off += B_DIM * M_DIM * 4;
    float* Tmp             = (float*)(ws + off);          off += B_DIM * 2048 * 4;

    const int CT = 256;
    auto blocks = [](int n) { return (n + 255) / 256; };

    // ---- phase 0: bf16 conversions + state init ----
    k_f2bf<<<blocks(ROWS_X * I_DIM), CT, 0, stream>>>(inputs,   Xbf,   ROWS_X * I_DIM);
    k_f2bf<<<blocks(ROWS_X * M_DIM), CT, 0, stream>>>(sememe_h, Sbf,   ROWS_X * M_DIM);
    k_f2bf<<<blocks(1536 * 512), CT, 0, stream>>>(W_ioux, Wioux, 1536 * 512);
    k_f2bf<<<blocks(512 * 512),  CT, 0, stream>>>(W_fx,   Wfx,   512 * 512);
    k_f2bf<<<blocks(512 * 512),  CT, 0, stream>>>(W_fxs,  Wfxs,  512 * 512);
    k_f2bf<<<blocks(1536 * 512), CT, 0, stream>>>(W_ious, Wious, 1536 * 512);
    k_f2bf<<<blocks(512 * 512),  CT, 0, stream>>>(W_fs,   Wfs,   512 * 512);
    k_f2bf<<<blocks(1536 * 512), CT, 0, stream>>>(W_iouh, Whcat,              1536 * 512);
    k_f2bf<<<blocks(512 * 512),  CT, 0, stream>>>(W_fh,   Whcat + 1536 * 512, 512 * 512);
    k_f2bf<<<blocks(B_DIM * M_DIM), CT, 0, stream>>>(h0, Hbf, B_DIM * M_DIM);
    hipMemcpyAsync(Cst, c0, (size_t)B_DIM * M_DIM * sizeof(float),
                   hipMemcpyDeviceToDevice, stream);

    // ---- phase 1: biases into P, then 5 accumulate-GEMMs (time-parallel) ----
    {
        size_t total = (size_t)ROWS_X * PCOLS;
        k_init_P<<<(unsigned)((total + CT - 1) / CT), CT, 0, stream>>>(
            P, b_ioux, b_iouh, b_ious, b_fx, b_fh, b_fxs, b_fs);
    }
    dim3 blk(128);
    k_gemm_bf16<<<dim3(ROWS_X / 64, 1536 / 64), blk, 0, stream>>>(Xbf, Wioux, P, PCOLS, 0,    1);
    k_gemm_bf16<<<dim3(ROWS_X / 64,  512 / 64), blk, 0, stream>>>(Xbf, Wfx,   P, PCOLS, 1536, 1);
    k_gemm_bf16<<<dim3(ROWS_X / 64,  512 / 64), blk, 0, stream>>>(Xbf, Wfxs,  P, PCOLS, 2048, 1);
    k_gemm_bf16<<<dim3(ROWS_X / 64, 1536 / 64), blk, 0, stream>>>(Sbf, Wious, P, PCOLS, 0,    1);
    k_gemm_bf16<<<dim3(ROWS_X / 64,  512 / 64), blk, 0, stream>>>(Sbf, Wfs,   P, PCOLS, 2048, 1);

    // ---- phase 2: sequential scan (256 steps) ----
    for (int t = 0; t < T_DIM; ++t) {
        // tmp[64,2048] = h_bf16 @ [W_iouh; W_fh]^T   (overwrite, no accumulate)
        k_gemm_bf16<<<dim3(B_DIM / 64, 2048 / 64), blk, 0, stream>>>(
            Hbf, Whcat, Tmp, 2048, 0, 0);
        k_gates<<<blocks(B_DIM * M_DIM), CT, 0, stream>>>(
            P, Tmp, Cst, sememe_c, out, Hbf, t);
    }
}
